// NaViTBackbone_6339371729794
// MI455X (gfx1250) — matrix operand order repaired and checked by reading, not compiled, not run on hardware
//
#include <hip/hip_runtime.h>
#include <hip/hip_bf16.h>
#include <math.h>

// ---------------------------------------------------------------------------
// NaViT forward for MI455X (gfx1250, wave32, WMMA).
// bf16 WMMA (f32 accumulate) for all GEMMs and attention matmuls.
// ---------------------------------------------------------------------------

typedef __attribute__((ext_vector_type(16))) __bf16 v16bf;
typedef __attribute__((ext_vector_type(8)))  __bf16 v8bf;
typedef __attribute__((ext_vector_type(8)))  float  v8f;
typedef __attribute__((ext_vector_type(4)))  float  f32x4;

#define DMODEL   1024
#define NHEAD    16
#define DHEAD    64
#define SEQ      640
#define BATCH    4
#define NIMG     18
#define QB_TILES 10      // SEQ / 64
#define PDIM     3072
#define MLPDIM   2048
#define NCLS     1000

__device__ __forceinline__ __bf16 tobf(float f) { return (__bf16)f; }

__device__ __forceinline__ v16bf cat16(v8bf lo, v8bf hi) {
  return __builtin_shufflevector(lo, hi, 0, 1, 2, 3, 4, 5, 6, 7,
                                         8, 9, 10, 11, 12, 13, 14, 15);
}

// ------------------------------ LayerNorm ----------------------------------
__global__ void ln_kernel(const float* __restrict__ X, float* __restrict__ Y,
                          const float* __restrict__ g, const float* __restrict__ bta,
                          int C)
{
  const int row = blockIdx.x;
  const float* x = X + (size_t)row * C;
  float* y = Y + (size_t)row * C;
  __shared__ float red[256];
  const int tid = threadIdx.x;
  float s = 0.f;
  for (int i = tid; i < C; i += 256) s += x[i];
  red[tid] = s; __syncthreads();
  for (int st = 128; st > 0; st >>= 1) { if (tid < st) red[tid] += red[tid + st]; __syncthreads(); }
  const float mean = red[0] / (float)C;
  __syncthreads();
  float v = 0.f;
  for (int i = tid; i < C; i += 256) { float d = x[i] - mean; v += d * d; }
  red[tid] = v; __syncthreads();
  for (int st = 128; st > 0; st >>= 1) { if (tid < st) red[tid] += red[tid + st]; __syncthreads(); }
  const float rstd = rsqrtf(red[0] / (float)C + 1e-5f);
  for (int i = tid; i < C; i += 256)
    y[i] = (x[i] - mean) * rstd * g[i] + bta[i];
}

// ------------------------- per-head LayerNorm (C=64) -----------------------
__global__ void headln_kernel(float* __restrict__ buf,
                              const float* __restrict__ g, const float* __restrict__ bta,
                              int tok_stride, int heads)
{
  const int rid = blockIdx.x;
  const int tok = rid / heads, h = rid % heads;
  float* x = buf + (size_t)tok * tok_stride + h * DHEAD;
  __shared__ float red[64];
  const int t = threadIdx.x;
  const float v = x[t];
  red[t] = v; __syncthreads();
  for (int st = 32; st > 0; st >>= 1) { if (t < st) red[t] += red[t + st]; __syncthreads(); }
  const float mean = red[0] / 64.f;
  __syncthreads();
  const float d = v - mean;
  red[t] = d * d; __syncthreads();
  for (int st = 32; st > 0; st >>= 1) { if (t < st) red[t] += red[t + st]; __syncthreads(); }
  const float rstd = rsqrtf(red[0] / 64.f + 1e-5f);
  x[t] = d * rstd * g[t] + bta[t];
}

// ----------------------- factorized 2D pos-embed add -----------------------
__global__ void posadd_kernel(float* __restrict__ x,
                              const float* __restrict__ ph, const float* __restrict__ pw,
                              const int* __restrict__ hi, const int* __restrict__ wi)
{
  const int tok = blockIdx.x;
  const int a = hi[tok], b = wi[tok];
  float* xr = x + (size_t)tok * DMODEL;
  const float* pa = ph + (size_t)a * DMODEL;
  const float* pb = pw + (size_t)b * DMODEL;
  for (int d = threadIdx.x; d < DMODEL; d += 256) xr[d] += pa[d] + pb[d];
}

#define LDSA  40   // row stride (elements) for As tiles, 80B: 16B aligned
#define LDSBP 40   // row stride (elements) for Bp tiles (transposed [n][k])

// --------------------------- WMMA GEMM (fast) ------------------------------
// Full-tile path: M % 64 == 0, N % 128 == 0, K % 32 == 0.  No bounds checks.
// Block tile 64x128, 8 waves, wave = 16x64 strip (4 accumulators -> A frag
// reused 4x).  Double-buffered LDS: one s_barrier per K step; next tile's
// global loads are issued before the WMMAs so they overlap compute.
template <bool BIAS, bool RES, int ACT>
__global__ void gemm_fast_kernel(const float* __restrict__ A, int lda,
                                 const float* __restrict__ Bm, int ldb,
                                 float* __restrict__ C, int ldc,
                                 const float* __restrict__ bias,
                                 const float* __restrict__ res, int ldres,
                                 int K)
{
  __shared__ __bf16 As[2][64 * LDSA];
  __shared__ __bf16 Bp[2][128 * LDSBP];
  const int tid = threadIdx.x;
  const int lane = tid & 31, wave = tid >> 5;
  const int wr = (wave & 3) * 16, wc = (wave >> 2) * 64;
  const int m0 = blockIdx.y * 64, n0 = blockIdx.x * 128;

  // staging coords: A 64x32 (8 elems/thread), B 32x128 (16 elems/thread)
  const int arow = tid >> 2;           // 0..63
  const int acol = (tid & 3) * 8;      // 0,8,16,24
  const int bn   = tid & 127;          // 0..127
  const int bk   = (tid >> 7) * 16;    // 0,16
  const float* gA = A + (size_t)(m0 + arow) * lda + acol;
  const float* gB = Bm + (size_t)bk * ldb + n0 + bn;

  // fragment coords
  const int ar = wr + (lane & 15);
  const int ak = (lane >> 4) * 16;
  const int kb = (lane >> 4) * 16;
  const int bcol = wc + (lane & 15);

  f32x4 ra0, ra1;
  float rb[16];

  auto gload = [&](int k0) {
    ra0 = *(const f32x4*)(gA + k0);
    ra1 = *(const f32x4*)(gA + k0 + 4);
    const float* gb = gB + (size_t)k0 * ldb;
#pragma unroll
    for (int j = 0; j < 16; ++j) rb[j] = gb[(size_t)j * ldb];
  };
  auto sstore = [&](int buf) {
    v8bf av;
#pragma unroll
    for (int j = 0; j < 4; ++j) { av[j] = tobf(ra0[j]); av[j + 4] = tobf(ra1[j]); }
    *(v8bf*)&As[buf][arow * LDSA + acol] = av;
    v8bf b0, b1;
#pragma unroll
    for (int j = 0; j < 8; ++j) { b0[j] = tobf(rb[j]); b1[j] = tobf(rb[j + 8]); }
    *(v8bf*)&Bp[buf][bn * LDSBP + bk] = b0;
    *(v8bf*)&Bp[buf][bn * LDSBP + bk + 8] = b1;
  };

  v8f acc[4] = {};
  auto compute = [&](int buf) {
    const v16bf a = cat16(*(const v8bf*)&As[buf][ar * LDSA + ak],
                          *(const v8bf*)&As[buf][ar * LDSA + ak + 8]);
#pragma unroll
    for (int c = 0; c < 4; ++c) {
      const int bo = (bcol + c * 16) * LDSBP + kb;
      const v16bf bfr = cat16(*(const v8bf*)&Bp[buf][bo],
                              *(const v8bf*)&Bp[buf][bo + 8]);
      acc[c] = __builtin_amdgcn_wmma_f32_16x16x32_bf16(false, a, false, bfr,
                                                       (short)0, acc[c], false, false);
    }
  };

  gload(0);
  sstore(0);
  __syncthreads();
  int buf = 0;
  for (int k0 = 32; k0 < K; k0 += 32) {
    gload(k0);          // issue next tile's global loads
    compute(buf);       // WMMAs overlap the loads
    sstore(buf ^ 1);    // waits loads, writes other buffer
    __syncthreads();    // single barrier per K step
    buf ^= 1;
  }
  compute(buf);

  // epilogue (branch-free via template params)
  const int crow = wr + (lane >> 4) * 8;
  const int ccol = wc + (lane & 15);
  float bs[4] = {0.f, 0.f, 0.f, 0.f};
  if (BIAS) {
#pragma unroll
    for (int c = 0; c < 4; ++c) bs[c] = bias[n0 + ccol + c * 16];
  }
#pragma unroll
  for (int c = 0; c < 4; ++c) {
#pragma unroll
    for (int v = 0; v < 8; ++v) {
      const int gr = m0 + crow + v;
      const int gc = n0 + ccol + c * 16;
      float val = acc[c][v];
      if (BIAS) val += bs[c];
      if (ACT == 1) val = 0.5f * val * (1.f + erff(val * 0.70710678118654752f));
      if (RES) val += res[(size_t)gr * ldres + gc];
      C[(size_t)gr * ldc + gc] = val;
    }
  }
}

// --------------------------- WMMA GEMM (guarded) ---------------------------
// Bounds-tolerant 64x64 path for the small GEMMs (M=1, M=72, N=1000).
// Loads use CLAMPED indices everywhere (A row r only affects C row r; B col n
// only affects C col n; bias/res values for never-stored elements are
// irrelevant), so ALL loads are unconditional; only the final store is
// predicated.  Templated on <BIAS, RES> (no GELU user on this path).
template <bool BIAS, bool RES>
__global__ void gemm_guard_kernel(const float* __restrict__ A, int lda,
                                  const float* __restrict__ Bm, int ldb,
                                  float* __restrict__ C, int ldc,
                                  const float* __restrict__ bias,
                                  const float* __restrict__ res, int ldres,
                                  int M, int N, int K)
{
  __shared__ __bf16 As[64 * LDSA];
  __shared__ __bf16 Bp[64 * LDSBP];   // transposed [n][k]
  const int tid = threadIdx.x;
  const int lane = tid & 31, wave = tid >> 5;
  const int wr = (wave & 3) * 16, wc = (wave >> 2) * 32;
  const int m0 = blockIdx.y * 64, n0 = blockIdx.x * 64;
  const int arow = tid >> 2;
  const int acol = (tid & 3) * 8;
  const int bn   = tid & 63;
  const int bk   = (tid >> 6) * 8;
  const int ar = wr + (lane & 15);
  const int ak = (lane >> 4) * 16;
  const int kb = (lane >> 4) * 16;
  const int bcol = wc + (lane & 15);

  // clamped (always in-bounds) staging addresses
  const int arow_c = min(m0 + arow, M - 1);
  const int bn_c   = min(n0 + bn, N - 1);
  const float* gA = A + (size_t)arow_c * lda + acol;
  const float* gB = Bm + (size_t)bk * ldb + bn_c;

  v8f acc0 = {}, acc1 = {};
  for (int k0 = 0; k0 < K; k0 += 32) {
    const f32x4 fa0 = *(const f32x4*)(gA + k0);
    const f32x4 fa1 = *(const f32x4*)(gA + k0 + 4);
    v8bf av;
#pragma unroll
    for (int j = 0; j < 4; ++j) { av[j] = tobf(fa0[j]); av[j + 4] = tobf(fa1[j]); }
    *(v8bf*)&As[arow * LDSA + acol] = av;
    const float* gb = gB + (size_t)k0 * ldb;
    v8bf bv;
#pragma unroll
    for (int j = 0; j < 8; ++j) bv[j] = tobf(gb[(size_t)j * ldb]);
    *(v8bf*)&Bp[bn * LDSBP + bk] = bv;
    __syncthreads();
    const v16bf a  = cat16(*(const v8bf*)&As[ar * LDSA + ak],
                           *(const v8bf*)&As[ar * LDSA + ak + 8]);
    const v16bf b0 = cat16(*(const v8bf*)&Bp[bcol * LDSBP + kb],
                           *(const v8bf*)&Bp[bcol * LDSBP + kb + 8]);
    const v16bf b1 = cat16(*(const v8bf*)&Bp[(bcol + 16) * LDSBP + kb],
                           *(const v8bf*)&Bp[(bcol + 16) * LDSBP + kb + 8]);
    acc0 = __builtin_amdgcn_wmma_f32_16x16x32_bf16(false, a, false, b0, (short)0, acc0, false, false);
    acc1 = __builtin_amdgcn_wmma_f32_16x16x32_bf16(false, a, false, b1, (short)0, acc1, false, false);
    __syncthreads();
  }
  const int crow = wr + (lane >> 4) * 8;
  const int ccol = wc + (lane & 15);
  // hoisted, clamped bias loads (2 per lane)
  float bs0 = 0.f, bs1 = 0.f;
  if (BIAS) {
    bs0 = bias[min(n0 + ccol, N - 1)];
    bs1 = bias[min(n0 + ccol + 16, N - 1)];
  }
#pragma unroll
  for (int v = 0; v < 8; ++v) {
    const int gr  = m0 + crow + v;
    const int grc = min(gr, M - 1);
#pragma unroll
    for (int sub = 0; sub < 2; ++sub) {
      const int gc  = n0 + ccol + sub * 16;
      const int gcc = min(gc, N - 1);
      float val = sub ? acc1[v] : acc0[v];
      if (BIAS) val += sub ? bs1 : bs0;
      if (RES)  val += res[(size_t)grc * ldres + gcc];
      if (gr < M && gc < N) C[(size_t)gr * ldc + gc] = val;
    }
  }
}

// ------------------------------ Attention ----------------------------------
// One workgroup per (batch, head, 64-query block).  Full 64x640 score strip
// in LDS (~165 KB of the 320 KB WGP LDS).  WMMA for Q K^T and P V.
#define SROW 648   // 648*4B = 2592B row stride, 16B aligned
__global__ void attn_kernel(const float* __restrict__ Q,   // [B*SEQ, 1024]
                            const float* __restrict__ KV,  // [B*SEQ, 2048] (k|v)
                            const int* __restrict__ ids,   // [B, SEQ]
                            float* __restrict__ O)         // [B*SEQ, 1024]
{
  extern __shared__ char smem[];
  float* S = (float*)smem;                                     // [64][SROW]
  int* sid = (int*)(smem + (size_t)64 * SROW * sizeof(float)); // [SEQ]
  const int bid = blockIdx.x;
  const int qb = bid % QB_TILES;
  const int h  = (bid / QB_TILES) % NHEAD;
  const int b  = bid / (QB_TILES * NHEAD);
  const int tid = threadIdx.x, lane = tid & 31, wave = tid >> 5;

  for (int j = tid; j < SEQ; j += 256) sid[j] = ids[b * SEQ + j];
  __syncthreads();

  // ---- phase 1: S = mask(scale * Q K^T) ----
  {
    const int qs = wave & 3;
    const float* qrow = Q + ((size_t)(b * SEQ + qb * 64 + qs * 16 + (lane & 15))) * DMODEL + h * DHEAD;
    const int k0 = (lane >> 4) * 16;
    v16bf a0, a1;
    {
      const f32x4 f0 = *(const f32x4*)(qrow + k0);
      const f32x4 f1 = *(const f32x4*)(qrow + k0 + 4);
      const f32x4 f2 = *(const f32x4*)(qrow + k0 + 8);
      const f32x4 f3 = *(const f32x4*)(qrow + k0 + 12);
      const f32x4 f4 = *(const f32x4*)(qrow + 32 + k0);
      const f32x4 f5 = *(const f32x4*)(qrow + 32 + k0 + 4);
      const f32x4 f6 = *(const f32x4*)(qrow + 32 + k0 + 8);
      const f32x4 f7 = *(const f32x4*)(qrow + 32 + k0 + 12);
#pragma unroll
      for (int j = 0; j < 4; ++j) {
        a0[j] = tobf(f0[j]); a0[j + 4]  = tobf(f1[j]);
        a0[j + 8] = tobf(f2[j]); a0[j + 12] = tobf(f3[j]);
        a1[j] = tobf(f4[j]); a1[j + 4]  = tobf(f5[j]);
        a1[j + 8] = tobf(f6[j]); a1[j + 12] = tobf(f7[j]);
      }
    }
    const int d0 = (lane >> 4) * 16;
    for (int kt = (wave >> 2); kt < SEQ / 16; kt += 2) {
      const int key = kt * 16 + (lane & 15);
      const float* krow = KV + ((size_t)(b * SEQ + key)) * (2 * DMODEL) + h * DHEAD;
      v16bf b0, b1;
      {
        const f32x4 f0 = *(const f32x4*)(krow + d0);
        const f32x4 f1 = *(const f32x4*)(krow + d0 + 4);
        const f32x4 f2 = *(const f32x4*)(krow + d0 + 8);
        const f32x4 f3 = *(const f32x4*)(krow + d0 + 12);
        const f32x4 f4 = *(const f32x4*)(krow + 32 + d0);
        const f32x4 f5 = *(const f32x4*)(krow + 32 + d0 + 4);
        const f32x4 f6 = *(const f32x4*)(krow + 32 + d0 + 8);
        const f32x4 f7 = *(const f32x4*)(krow + 32 + d0 + 12);
#pragma unroll
        for (int j = 0; j < 4; ++j) {
          b0[j] = tobf(f0[j]); b0[j + 4]  = tobf(f1[j]);
          b0[j + 8] = tobf(f2[j]); b0[j + 12] = tobf(f3[j]);
          b1[j] = tobf(f4[j]); b1[j + 4]  = tobf(f5[j]);
          b1[j + 8] = tobf(f6[j]); b1[j + 12] = tobf(f7[j]);
        }
      }
      v8f acc = {};
      acc = __builtin_amdgcn_wmma_f32_16x16x32_bf16(false, a0, false, b0, (short)0, acc, false, false);
      acc = __builtin_amdgcn_wmma_f32_16x16x32_bf16(false, a1, false, b1, (short)0, acc, false, false);
      const int idk = sid[key];
      const int rb = qs * 16 + (lane >> 4) * 8;
#pragma unroll
      for (int v = 0; v < 8; ++v) {
        const int r = rb + v;
        const int idq = sid[qb * 64 + r];
        S[r * SROW + key] = (idq == idk) ? acc[v] * 0.125f : -1e30f;  // scale = 64^-0.5
      }
    }
  }
  __syncthreads();

  // ---- phase 2: row softmax ----
  if (tid < 64) {
    float* srow = S + tid * SROW;
    float mx = -1e30f;
    for (int j = 0; j < SEQ; ++j) mx = fmaxf(mx, srow[j]);
    float sum = 0.f;
    for (int j = 0; j < SEQ; ++j) { const float e = __expf(srow[j] - mx); srow[j] = e; sum += e; }
    const float inv = 1.f / sum;
    for (int j = 0; j < SEQ; ++j) srow[j] *= inv;
  }
  __syncthreads();

  // ---- phase 3: O = P V ----
  {
    const int qs = wave & 3;
    const int r = qs * 16 + (lane & 15);
    const int k0 = (lane >> 4) * 16;
    const int kb0 = (lane >> 4) * 16;
    for (int ds = (wave >> 2); ds < 4; ds += 2) {
      const float* vbase = KV + ((size_t)(b * SEQ)) * (2 * DMODEL) + DMODEL + h * DHEAD
                           + ds * 16 + (lane & 15);
      v8f acc = {};
      for (int kt = 0; kt < SEQ; kt += 32) {
        const float* srow = S + r * SROW + kt;
        v16bf a;
        {
          const f32x4 f0 = *(const f32x4*)(srow + k0);
          const f32x4 f1 = *(const f32x4*)(srow + k0 + 4);
          const f32x4 f2 = *(const f32x4*)(srow + k0 + 8);
          const f32x4 f3 = *(const f32x4*)(srow + k0 + 12);
#pragma unroll
          for (int j = 0; j < 4; ++j) {
            a[j] = tobf(f0[j]); a[j + 4]  = tobf(f1[j]);
            a[j + 8] = tobf(f2[j]); a[j + 12] = tobf(f3[j]);
          }
        }
        v16bf bb;   // V gather: 16 lane-coalesced strided loads
#pragma unroll
        for (int e = 0; e < 16; ++e)
          bb[e] = tobf(vbase[(size_t)(kt + kb0 + e) * (2 * DMODEL)]);
        acc = __builtin_amdgcn_wmma_f32_16x16x32_bf16(false, a, false, bb, (short)0, acc, false, false);
      }
      const int rb = qs * 16 + (lane >> 4) * 8;
#pragma unroll
      for (int v = 0; v < 8; ++v)
        O[((size_t)(b * SEQ + qb * 64 + rb + v)) * DMODEL + h * DHEAD + ds * 16 + (lane & 15)] = acc[v];
    }
  }
}

// --------------------------- attention pooling -----------------------------
__global__ void poolattn_kernel(const float* __restrict__ Qrow, // [1024] shared query
                                const float* __restrict__ KV,   // [B*SEQ, 2048]
                                const int* __restrict__ ids,
                                float* __restrict__ Out)        // [B*M, 1024]
{
  const int bid = blockIdx.x;
  const int h = bid % NHEAD;
  const int m = (bid / NHEAD) % NIMG;
  const int b = bid / (NHEAD * NIMG);
  __shared__ float sc[SEQ];
  __shared__ float qv[64];
  __shared__ float red[64];
  const int t = threadIdx.x;
  qv[t] = Qrow[h * DHEAD + t];
  __syncthreads();
  for (int j = t; j < SEQ; j += 64) {
    float s = -1e30f;
    if (ids[b * SEQ + j] == m) {
      const float* kr = KV + (size_t)(b * SEQ + j) * (2 * DMODEL) + h * DHEAD;
      float d = 0.f;
      for (int e = 0; e < 64; ++e) d += qv[e] * kr[e];
      s = d * 0.125f;
    }
    sc[j] = s;
  }
  __syncthreads();
  float mx = -1e30f;
  for (int j = t; j < SEQ; j += 64) mx = fmaxf(mx, sc[j]);
  red[t] = mx; __syncthreads();
  for (int st = 32; st > 0; st >>= 1) { if (t < st) red[t] = fmaxf(red[t], red[t + st]); __syncthreads(); }
  mx = red[0]; __syncthreads();
  float ls = 0.f;
  for (int j = t; j < SEQ; j += 64) { const float e = __expf(sc[j] - mx); sc[j] = e; ls += e; }
  red[t] = ls; __syncthreads();
  for (int st = 32; st > 0; st >>= 1) { if (t < st) red[t] += red[t + st]; __syncthreads(); }
  const float inv = 1.f / red[0];
  __syncthreads();
  const float* vb = KV + (size_t)(b * SEQ) * (2 * DMODEL) + DMODEL + h * DHEAD + t;
  float o = 0.f;
  for (int j = 0; j < SEQ; ++j) o += sc[j] * vb[(size_t)j * (2 * DMODEL)];
  Out[(size_t)(b * NIMG + m) * DMODEL + h * DHEAD + t] = o * inv;
}

// ---------------------------------------------------------------------------
extern "C" void kernel_launch(void* const* d_in, const int* in_sizes, int n_in,
                              void* d_out, int out_size, void* d_ws, size_t ws_size,
                              hipStream_t stream)
{
  (void)in_sizes; (void)n_in; (void)out_size; (void)ws_size;
  const float* patches  = (const float*)d_in[0];
  const float* pe_ln1_g = (const float*)d_in[1];
  const float* pe_ln1_b = (const float*)d_in[2];
  const float* W_pe     = (const float*)d_in[3];
  const float* b_pe     = (const float*)d_in[4];
  const float* pe_ln2_g = (const float*)d_in[5];
  const float* pe_ln2_b = (const float*)d_in[6];
  const float* pos_h    = (const float*)d_in[7];
  const float* pos_w    = (const float*)d_in[8];
  const float* ln1_g    = (const float*)d_in[9];
  const float* ln1_b    = (const float*)d_in[10];
  const float* qn_g     = (const float*)d_in[11];
  const float* qn_b     = (const float*)d_in[12];
  const float* kn_g     = (const float*)d_in[13];
  const float* kn_b     = (const float*)d_in[14];
  const float* Wq       = (const float*)d_in[15];
  const float* Wkv      = (const float*)d_in[16];
  const float* Wo       = (const float*)d_in[17];
  const float* ln2_g    = (const float*)d_in[18];
  const float* ln2_b    = (const float*)d_in[19];
  const float* W1       = (const float*)d_in[20];
  const float* b1       = (const float*)d_in[21];
  const float* W2       = (const float*)d_in[22];
  const float* b2       = (const float*)d_in[23];
  const float* fin_g    = (const float*)d_in[24];
  const float* fin_b    = (const float*)d_in[25];
  const float* pool_q   = (const float*)d_in[26];
  const float* pl_ln_g  = (const float*)d_in[27];
  const float* pl_ln_b  = (const float*)d_in[28];
  const float* pl_qn_g  = (const float*)d_in[29];
  const float* pl_qn_b  = (const float*)d_in[30];
  const float* pl_kn_g  = (const float*)d_in[31];
  const float* pl_kn_b  = (const float*)d_in[32];
  const float* pl_Wq    = (const float*)d_in[33];
  const float* pl_Wkv   = (const float*)d_in[34];
  const float* pl_Wo    = (const float*)d_in[35];
  const float* head_g   = (const float*)d_in[36];
  const float* head_b   = (const float*)d_in[37];
  const float* W_head   = (const float*)d_in[38];
  const float* b_head   = (const float*)d_in[39];
  const int*   h_idx    = (const int*)d_in[40];
  const int*   w_idx    = (const int*)d_in[41];
  const int*   img_ids  = (const int*)d_in[42];

  const int R = BATCH * SEQ;          // 2560 token rows
  float* ws = (float*)d_ws;
  float* x         = ws;                 // [R,1024]
  float* xn        = x   + (size_t)R * DMODEL;
  float* q         = xn  + (size_t)R * DMODEL;
  float* kv        = q   + (size_t)R * DMODEL;     // [R,2048] (kv & mlp hidden)
  float* ao        = kv  + (size_t)R * 2 * DMODEL; // [R,1024]
  float* pln       = ao  + (size_t)R * DMODEL;     // [R,3072]
  float* pqn       = pln + (size_t)R * PDIM;       // [1024]
  float* pqp       = pqn + DMODEL;                 // [1024]
  float* ph_buf    = pqp + DMODEL;                 // [72,1024]
  float* pooled    = ph_buf + (size_t)BATCH * NIMG * DMODEL;
  float* pooled_ln = pooled + (size_t)BATCH * NIMG * DMODEL;

  auto GEMM = [&](const float* A, int lda, const float* W, int ldb, float* Cc, int ldc,
                  const float* bias, const float* res, int ldres,
                  int M, int N, int K, int act) {
    if ((M & 63) == 0 && (N & 127) == 0) {
      dim3 grid(N / 128, M / 64);
      if (!bias && !res)
        gemm_fast_kernel<false, false, 0><<<grid, 256, 0, stream>>>(A, lda, W, ldb, Cc, ldc, bias, res, ldres, K);
      else if (bias && !res && act == 0)
        gemm_fast_kernel<true, false, 0><<<grid, 256, 0, stream>>>(A, lda, W, ldb, Cc, ldc, bias, res, ldres, K);
      else if (bias && !res)
        gemm_fast_kernel<true, false, 1><<<grid, 256, 0, stream>>>(A, lda, W, ldb, Cc, ldc, bias, res, ldres, K);
      else if (!bias && res)
        gemm_fast_kernel<false, true, 0><<<grid, 256, 0, stream>>>(A, lda, W, ldb, Cc, ldc, bias, res, ldres, K);
      else
        gemm_fast_kernel<true, true, 0><<<grid, 256, 0, stream>>>(A, lda, W, ldb, Cc, ldc, bias, res, ldres, K);
    } else {
      dim3 grid((N + 63) / 64, (M + 63) / 64);
      if (bias && !res)
        gemm_guard_kernel<true, false><<<grid, 256, 0, stream>>>(A, lda, W, ldb, Cc, ldc, bias, res, ldres, M, N, K);
      else if (!bias && res)
        gemm_guard_kernel<false, true><<<grid, 256, 0, stream>>>(A, lda, W, ldb, Cc, ldc, bias, res, ldres, M, N, K);
      else if (bias && res)
        gemm_guard_kernel<true, true><<<grid, 256, 0, stream>>>(A, lda, W, ldb, Cc, ldc, bias, res, ldres, M, N, K);
      else
        gemm_guard_kernel<false, false><<<grid, 256, 0, stream>>>(A, lda, W, ldb, Cc, ldc, bias, res, ldres, M, N, K);
    }
  };

  // ---- patch embedding: LN -> Linear -> LN -> +pos ----
  ln_kernel<<<R, 256, 0, stream>>>(patches, pln, pe_ln1_g, pe_ln1_b, PDIM);
  GEMM(pln, PDIM, W_pe, DMODEL, q, DMODEL, b_pe, nullptr, 0, R, DMODEL, PDIM, 0);
  ln_kernel<<<R, 256, 0, stream>>>(q, x, pe_ln2_g, pe_ln2_b, DMODEL);
  posadd_kernel<<<R, 256, 0, stream>>>(x, pos_h, pos_w, h_idx, w_idx);

  const size_t attn_smem = (size_t)64 * SROW * sizeof(float) + (size_t)SEQ * sizeof(int);
  (void)hipFuncSetAttribute((const void*)attn_kernel,
                            hipFuncAttributeMaxDynamicSharedMemorySize, (int)attn_smem);

  // ---- transformer layers ----
  for (int l = 0; l < 6; ++l) {
    const float* Wq_l  = Wq  + (size_t)l * DMODEL * DMODEL;
    const float* Wkv_l = Wkv + (size_t)l * DMODEL * 2 * DMODEL;
    const float* Wo_l  = Wo  + (size_t)l * DMODEL * DMODEL;
    const float* W1_l  = W1  + (size_t)l * DMODEL * MLPDIM;
    const float* W2_l  = W2  + (size_t)l * MLPDIM * DMODEL;

    ln_kernel<<<R, 256, 0, stream>>>(x, xn, ln1_g + l * DMODEL, ln1_b + l * DMODEL, DMODEL);
    GEMM(xn, DMODEL, Wq_l, DMODEL, q, DMODEL, nullptr, nullptr, 0, R, DMODEL, DMODEL, 0);
    headln_kernel<<<R * NHEAD, 64, 0, stream>>>(q, qn_g + l * DHEAD, qn_b + l * DHEAD, DMODEL, NHEAD);
    GEMM(xn, DMODEL, Wkv_l, 2 * DMODEL, kv, 2 * DMODEL, nullptr, nullptr, 0, R, 2 * DMODEL, DMODEL, 0);
    headln_kernel<<<R * NHEAD, 64, 0, stream>>>(kv, kn_g + l * DHEAD, kn_b + l * DHEAD, 2 * DMODEL, NHEAD);
    attn_kernel<<<BATCH * NHEAD * QB_TILES, 256, attn_smem, stream>>>(q, kv, img_ids, ao);
    GEMM(ao, DMODEL, Wo_l, DMODEL, x, DMODEL, nullptr, x, DMODEL, R, DMODEL, DMODEL, 0);
    ln_kernel<<<R, 256, 0, stream>>>(x, xn, ln2_g + l * DMODEL, ln2_b + l * DMODEL, DMODEL);
    GEMM(xn, DMODEL, W1_l, MLPDIM, kv, MLPDIM, b1 + l * MLPDIM, nullptr, 0, R, MLPDIM, DMODEL, 1);
    GEMM(kv, MLPDIM, W2_l, DMODEL, x, DMODEL, b2 + l * DMODEL, x, DMODEL, R, DMODEL, MLPDIM, 0);
  }

  // ---- final LN + attention pooling + head ----
  ln_kernel<<<R, 256, 0, stream>>>(x, xn, fin_g, fin_b, DMODEL);
  ln_kernel<<<1, 256, 0, stream>>>(pool_q, pqn, pl_ln_g, pl_ln_b, DMODEL);
  GEMM(pqn, DMODEL, pl_Wq, DMODEL, pqp, DMODEL, nullptr, nullptr, 0, 1, DMODEL, DMODEL, 0);
  headln_kernel<<<NHEAD, 64, 0, stream>>>(pqp, pl_qn_g, pl_qn_b, DMODEL, NHEAD);
  GEMM(xn, DMODEL, pl_Wkv, 2 * DMODEL, kv, 2 * DMODEL, nullptr, nullptr, 0, R, 2 * DMODEL, DMODEL, 0);
  headln_kernel<<<R * NHEAD, 64, 0, stream>>>(kv, pl_kn_g, pl_kn_b, 2 * DMODEL, NHEAD);
  poolattn_kernel<<<BATCH * NIMG * NHEAD, 64, 0, stream>>>(pqp, kv, img_ids, ph_buf);
  GEMM(ph_buf, DMODEL, pl_Wo, DMODEL, pooled, DMODEL, nullptr, pool_q, 0,
       BATCH * NIMG, DMODEL, DMODEL, 0);
  ln_kernel<<<BATCH * NIMG, 256, 0, stream>>>(pooled, pooled_ln, head_g, head_b, DMODEL);
  GEMM(pooled_ln, DMODEL, W_head, NCLS, (float*)d_out, NCLS, b_head, nullptr, 0,
       BATCH * NIMG, NCLS, DMODEL, 0);
}
